// Model_24747601559963
// MI455X (gfx1250) — matrix operand order repaired and checked by reading, not compiled, not run on hardware
//
#include <hip/hip_runtime.h>
#include <hip/hip_bf16.h>
#include <math.h>

typedef __attribute__((ext_vector_type(16))) _Float16 v16h;
typedef __attribute__((ext_vector_type(8)))  _Float16 v8h;
typedef __attribute__((ext_vector_type(8)))  float    v8f;

#define Bv   32
#define Nn   64
#define FINc 64
#define FCc  128
#define UDc  128
#define NHc  4
#define LRELU_ALPHA 0.2f
#define ATT_NORM 0.08838834764831845f   // 1/sqrt(128)

enum { ACT_NONE = 0, ACT_RELU = 1, ACT_TANH = 2 };

// ---------------------------------------------------------------------------
// WMMA GEMM with transposed-B operand:
//   C[M,N] = act(A[M,K] @ B[K,N] + bias[N]),  B supplied as Bt[N,K] row-major.
// A,Bt f16 row-major, C f32. 4 waves/block; each wave owns one 16x16 C tile.
// Per k-step each lane does 2x b128 loads for A and 2x b128 loads for Bt
// (all 16B-aligned contiguous chunks), then one v_wmma_f32_16x16x32_f16.
// K%32==0, M%16==0, N%64==0; lda/ldb multiples of 8.
// Batched via grid.z with element strides sA/sB/sC (0 => shared operand).
// ---------------------------------------------------------------------------
__global__ void wmma_gemm(const _Float16* __restrict__ A,
                          const _Float16* __restrict__ Bt,
                          const float* __restrict__ bias,
                          float* __restrict__ C,
                          int M, int N, int K, int lda, int ldb, int ldc,
                          long sA, long sB, long sC, int act)
{
    (void)N;
    const long z = blockIdx.z;
    A  += z * sA;
    Bt += z * sB;
    C  += z * sC;
    const int tm   = blockIdx.y << 4;
    const int tn   = ((blockIdx.x << 2) + threadIdx.y) << 4;
    const int lane = threadIdx.x & 31;
    const int hlf  = lane >> 4;     // wave half selects K sub-range
    const int idx  = lane & 15;     // M row (A) / N col (Bt, C)

    v8f acc = {0.f, 0.f, 0.f, 0.f, 0.f, 0.f, 0.f, 0.f};

    const _Float16* Arow = A  + (long)(tm + idx) * lda + (hlf << 3);
    const _Float16* Brow = Bt + (long)(tn + idx) * ldb + (hlf << 3);

    for (int k0 = 0; k0 < K; k0 += 32) {
        // A 16x32 fragment (ISA 7.12.2): a[0..7]=A[m, k0+8h .. +7], a[8..15]=A[m, k0+8h+16 ..]
        v8h alo = *(const v8h*)(Arow + k0);
        v8h ahi = *(const v8h*)(Arow + k0 + 16);
        // B 32x16 fragment: lane idx = col n; b[0..7]=Bt[n, k0+8h .. +7], b[8..15]=Bt[n, k0+8h+16 ..]
        v8h blo = *(const v8h*)(Brow + k0);
        v8h bhi = *(const v8h*)(Brow + k0 + 16);
        v16h a = __builtin_shufflevector(alo, ahi, 0,1,2,3,4,5,6,7,8,9,10,11,12,13,14,15);
        v16h b = __builtin_shufflevector(blo, bhi, 0,1,2,3,4,5,6,7,8,9,10,11,12,13,14,15);
        acc = __builtin_amdgcn_wmma_f32_16x16x32_f16(false, a, false, b,
                                                     (short)0, acc, false, false);
    }

    const float bvl = bias ? bias[tn + idx] : 0.f;
    #pragma unroll
    for (int v = 0; v < 8; ++v) {
        float x = acc[v] + bvl;
        if (act == ACT_RELU)      x = fmaxf(x, 0.f);
        else if (act == ACT_TANH) x = tanhf(x);
        // C/D layout: VGPR v holds row M = 8*half + v, col = lane idx
        C[(long)(tm + (hlf << 3) + v) * ldc + (tn + idx)] = x;
    }
}

// ---------------------------------------------------------------------------
// Utility kernels
// ---------------------------------------------------------------------------
__global__ void k_cast(_Float16* __restrict__ dst, const float* __restrict__ src, long n) {
    long i = (long)blockIdx.x * 256 + threadIdx.x;
    if (i < n) dst[i] = (_Float16)src[i];
}

// Transpose + cast + K-pad: src f32 [K,N] -> dst f16 [N,Kp] (zero tail K..Kp).
// Batched via grid.z with element strides sS/sD.
__global__ void k_wtrans(_Float16* __restrict__ dst, const float* __restrict__ src,
                         int K, int N, int Kp, long sS, long sD) {
    const long z = blockIdx.z;
    src += z * sS;
    dst += z * sD;
    long i = (long)blockIdx.x * 256 + threadIdx.x;      // over N*Kp
    if (i >= (long)N * Kp) return;
    int n = (int)(i / Kp), k = (int)(i % Kp);
    dst[i] = (k < K) ? (_Float16)src[(long)k * N + n] : (_Float16)0.f;
}

// Cast + row pad: src f32 [rows,cs] -> dst f16 [rows,cd] (zero tail cols)
__global__ void k_cast_pad2d(_Float16* __restrict__ dst, const float* __restrict__ src,
                             int rows, int cs, int cd) {
    long i = (long)blockIdx.x * 256 + threadIdx.x;
    if (i >= (long)rows * cd) return;
    int r = (int)(i / cd), c = (int)(i % cd);
    dst[i] = (c < cs) ? (_Float16)src[(long)r * cs + c] : (_Float16)0.f;
}

__global__ void k_fill_h(_Float16* __restrict__ dst, long n) {
    long i = (long)blockIdx.x * 256 + threadIdx.x;
    if (i < n) dst[i] = (_Float16)0.f;
}

// es/ed: per-node source/dest attention scores.  h: [H, B*N, FC] f32
__global__ void k_gat_scores(const float* __restrict__ h,
                             const float* __restrict__ as_,
                             const float* __restrict__ ad_,
                             float* __restrict__ es, float* __restrict__ ed) {
    int i = blockIdx.x * 256 + threadIdx.x;            // over H*B*N = 8192
    if (i >= NHc * Bv * Nn) return;
    int hh = i >> 11;                                   // / (B*N)
    const float* hv = h + (long)i * FCc;
    float s = 0.f, d = 0.f;
    for (int o = 0; o < FCc; ++o) {
        float x = hv[o];
        s += x * as_[hh * FCc + o];
        d += x * ad_[hh * FCc + o];
    }
    es[i] = s; ed[i] = d;
}

// grid (N, B, H), block 64 (j).  att f16 [H,B,N,N]
__global__ void k_gat_softmax(const float* __restrict__ es, const float* __restrict__ ed,
                              const int* __restrict__ adj, _Float16* __restrict__ att) {
    const int i = blockIdx.x, b = blockIdx.y, h = blockIdx.z;
    const int j = threadIdx.x;
    __shared__ float sh[64];
    const long base = ((long)h * Bv + b) * Nn;
    float x = es[base + i] + ed[base + j];
    x = (x > 0.f) ? x : LRELU_ALPHA * x;
    const int a = adj[((long)b * Nn + i) * Nn + j];
    float e = (a > 0) ? x : -1e9f;
    sh[j] = e; __syncthreads();
    for (int s = 32; s > 0; s >>= 1) { if (j < s) sh[j] = fmaxf(sh[j], sh[j + s]); __syncthreads(); }
    const float mx = sh[0]; __syncthreads();
    const float ex = expf(e - mx);
    sh[j] = ex; __syncthreads();
    for (int s = 32; s > 0; s >>= 1) { if (j < s) sh[j] += sh[j + s]; __syncthreads(); }
    att[(base + i) * Nn + j] = (_Float16)(ex / sh[0]);
}

// out[b,n,o] = elu(mean_h ghx[h,b,n,o]);  total = B*N*FC
__global__ void k_mean_elu(const float* __restrict__ ghx, float* __restrict__ out, int total) {
    int i = blockIdx.x * 256 + threadIdx.x;
    if (i >= total) return;
    float s = 0.f;
    for (int h = 0; h < NHc; ++h) s += ghx[(long)h * total + i];
    s *= (1.0f / NHc);
    out[i] = (s > 0.f) ? s : (expf(s) - 1.0f);
}

// out[b,f] = sum_n in[b,n,f]
__global__ void k_sum_n(const float* __restrict__ in, float* __restrict__ out,
                        int B_, int N_, int F_) {
    int i = blockIdx.x * 256 + threadIdx.x;
    if (i >= B_ * F_) return;
    int b = i / F_, f = i % F_;
    float s = 0.f;
    for (int n = 0; n < N_; ++n) s += in[((long)b * N_ + n) * F_ + f];
    out[i] = s;
}

// 3DGCN pairwise message construction + adjacency-masked reduction over j.
// Pair biases are pre-folded into the a_* operands by the projection GEMMs.
__global__ void k_pair(const float* __restrict__ a_ss, const float* __restrict__ c_ss,
                       const float* __restrict__ a_vs, const float* __restrict__ c_vs,
                       const float* __restrict__ a_sv, const float* __restrict__ c_sv,
                       const float* __restrict__ a_vv, const float* __restrict__ c_vv,
                       const int* __restrict__ adj, const float* __restrict__ dists,
                       float* __restrict__ s_sum, float* __restrict__ v_sum) {
    const int bi = blockIdx.x;                 // b*N + i
    const int b  = bi >> 6;
    const int o  = threadIdx.x;                // channel 0..127
    const long rowS = (long)bi * FCc + o;
    const long rowV = (long)bi * 3 * FCc + o;
    const float ass  = a_ss[rowS];
    const float asv  = a_sv[rowS];
    const float avs0 = a_vs[rowV], avs1 = a_vs[rowV + FCc], avs2 = a_vs[rowV + 2 * FCc];
    const float avv0 = a_vv[rowV], avv1 = a_vv[rowV + FCc], avv2 = a_vv[rowV + 2 * FCc];
    float s0 = 0.f, s1 = 0.f;
    float v00 = 0.f, v01 = 0.f, v02 = 0.f, v10 = 0.f, v11 = 0.f, v12 = 0.f;
    const int*   arow = adj   + (long)bi * Nn;
    const float* drow = dists + (long)bi * Nn * 3;
    const long bbase = (long)b * Nn;
    for (int j = 0; j < Nn; ++j) {
        if (arow[j] == 0) continue;            // adj in {0,1}: masked term is exactly 0
        const float d0 = drow[3 * j], d1 = drow[3 * j + 1], d2 = drow[3 * j + 2];
        const long js = (bbase + j) * FCc + o;
        const long jv = (bbase + j) * 3 * FCc + o;
        s0 += fmaxf(ass + c_ss[js], 0.f);
        const float p0 = avs0 + c_vs[jv], p1 = avs1 + c_vs[jv + FCc], p2 = avs2 + c_vs[jv + 2 * FCc];
        s1 += fmaxf(p0 * d0 + p1 * d1 + p2 * d2, 0.f);
        const float ps = asv + c_sv[js];
        v00 += tanhf(ps * d0); v01 += tanhf(ps * d1); v02 += tanhf(ps * d2);
        v10 += tanhf(avv0 + c_vv[jv]);
        v11 += tanhf(avv1 + c_vv[jv + FCc]);
        v12 += tanhf(avv2 + c_vv[jv + 2 * FCc]);
    }
    s_sum[(long)bi * 2 * FCc + o]       = s0;
    s_sum[(long)bi * 2 * FCc + FCc + o] = s1;
    float* vo = v_sum + (long)bi * 3 * 2 * FCc;
    vo[0 * 2 * FCc + o] = v00; vo[0 * 2 * FCc + FCc + o] = v10;
    vo[1 * 2 * FCc + o] = v01; vo[1 * 2 * FCc + FCc + o] = v11;
    vo[2 * 2 * FCc + o] = v02; vo[2 * 2 * FCc + FCc + o] = v12;
}

// dst[r, 0:ca]=A, dst[r, ca:ca+cb]=B  (f32 -> f16)
__global__ void k_concat2_cast(_Float16* __restrict__ dst, const float* __restrict__ A,
                               const float* __restrict__ Bp, int rows, int ca, int cb) {
    int cd = ca + cb;
    long i = (long)blockIdx.x * 256 + threadIdx.x;
    if (i >= (long)rows * cd) return;
    int r = (int)(i / cd), c = (int)(i % cd);
    dst[i] = (_Float16)((c < ca) ? A[(long)r * ca + c] : Bp[(long)r * cb + (c - ca)]);
}

// x = stack([a,b,c], axis=1): dst [rows,3,cols] f16
__global__ void k_stack3_cast(_Float16* __restrict__ dst, const float* __restrict__ a,
                              const float* __restrict__ b, const float* __restrict__ c,
                              int rows, int cols) {
    long i = (long)blockIdx.x * 256 + threadIdx.x;
    if (i >= (long)rows * 3 * cols) return;
    int r = (int)(i / (3 * cols));
    int rem = (int)(i % (3 * cols));
    int s = rem / cols, ci = rem % cols;
    const float* src = (s == 0) ? a : (s == 1) ? b : c;
    dst[i] = (_Float16)src[(long)r * cols + ci];
}

// 3-token self-attention head (scale AFTER softmax, per reference), fused relu.
__global__ void k_selfattn(const float* __restrict__ Q, const float* __restrict__ Km,
                           const float* __restrict__ V, float* __restrict__ o) {
    const int b = blockIdx.x;
    const int t = threadIdx.x;                  // 128
    __shared__ float att[3][3];
    __shared__ float qs[3][UDc], ks[3][UDc], vs[3][UDc];
    for (int s = 0; s < 3; ++s) {
        long idx = ((long)b * 3 + s) * UDc + t;
        qs[s][t] = Q[idx]; ks[s][t] = Km[idx]; vs[s][t] = V[idx];
    }
    __syncthreads();
    if (t < 9) {
        int i = t / 3, j = t % 3;
        float s = 0.f;
        for (int oo = 0; oo < UDc; ++oo) s += qs[i][oo] * ks[j][oo];
        att[i][j] = s;
    }
    __syncthreads();
    if (t < 3) {
        float m = fmaxf(att[t][0], fmaxf(att[t][1], att[t][2]));
        float e0 = expf(att[t][0] - m), e1 = expf(att[t][1] - m), e2 = expf(att[t][2] - m);
        float s = e0 + e1 + e2;
        att[t][0] = e0 / s * ATT_NORM;
        att[t][1] = e1 / s * ATT_NORM;
        att[t][2] = e2 / s * ATT_NORM;
    }
    __syncthreads();
    for (int i = 0; i < 3; ++i) {
        float r = att[i][0] * vs[0][t] + att[i][1] * vs[1][t] + att[i][2] * vs[2][t];
        o[(long)b * (3 * UDc) + i * UDc + t] = fmaxf(r, 0.f);
    }
}

// final projection [B,128] @ [128,1] + b
__global__ void k_out(const float* __restrict__ x, const float* __restrict__ w,
                      const float* __restrict__ bias, float* __restrict__ out) {
    const int b = blockIdx.x;
    const int t = threadIdx.x;                  // 128
    __shared__ float sh[128];
    sh[t] = x[(long)b * UDc + t] * w[t];
    __syncthreads();
    for (int s = 64; s > 0; s >>= 1) { if (t < s) sh[t] += sh[t + s]; __syncthreads(); }
    if (t == 0) out[b] = sh[0] + bias[0];
}

// ---------------------------------------------------------------------------
// Host orchestration
// ---------------------------------------------------------------------------
extern "C" void kernel_launch(void* const* d_in, const int* in_sizes, int n_in,
                              void* d_out, int out_size, void* d_ws, size_t ws_size,
                              hipStream_t stream) {
    (void)in_sizes; (void)n_in; (void)out_size; (void)ws_size;

    const float* atoms = (const float*)d_in[0];   // [32,64,64]
    const int*   adj   = (const int*)  d_in[1];   // [32,64,64]
    const float* dists = (const float*)d_in[2];   // [32,64,64,3]
    const float* fp    = (const float*)d_in[3];   // [32,1489]
    auto P = [&](int i) -> const float* { return (const float*)d_in[4 + i]; };
    // param indices (dict insertion order):
    // 0 fpn_w1 1 fpn_b1 2 fpn_w2 3 fpn_b2 4 tdim_W 5 tdim_as 6 tdim_ad
    // 7 fcg1_w 8 fcg1_b 9 fcg2_w 10 fcg2_b
    // 11..22 l0: ss_w,ss_b,vs_w,vs_b,sv_w,sv_b,vv_w,vv_b,cs_w,cs_b,cv_w,cv_b
    // 23..34 l1 (same order)
    // 35 gat_W 36 gat_as 37 gat_ad
    // 38 so1_w 39 so1_b 40 so2_w 41 so2_b 42 vo1_w 43 vo1_b 44 vo2_w 45 vo2_b
    // 46 fc1_w 47 fc1_b 48 fc2_w 49 fc2_b
    // 50 q_w 51 q_b 52 k_w 53 k_b 54 v_w 55 v_b 56 o1_w 57 o1_b 58 o2_w 59 o2_b
    // 60 out_w 61 out_b

    // ---- workspace arena ----
    char* base = (char*)d_ws;
    size_t off = 0;
    auto alloc = [&](size_t bytes) -> void* {
        void* p = base + off;
        off = (off + bytes + 255) & ~(size_t)255;
        return p;
    };
    auto fh = [&](size_t n) { return (_Float16*)alloc(n * sizeof(_Float16)); };
    auto ff = [&](size_t n) { return (float*)alloc(n * sizeof(float)); };

    const int BN = Bv * Nn;            // 2048
    const int BN3 = BN * 3;            // 6144
    const int KFP = 1504;              // 1489 padded to multiple of 32

    // f16 transposed weights: Bt layout [N, Kp]
    _Float16* w_fpn1 = fh((size_t)512 * KFP);
    _Float16* w_fpn2 = fh(128 * 512);
    _Float16* w_tdim = fh(4 * 128 * 64);
    _Float16* w_fcg1 = fh(128 * 128);
    _Float16* w_fcg2 = fh(128 * 128);
    _Float16* w_l[2][6];               // ss,vs,sv,vv (Bt [128,2fi]), cs,cv (Bt [128,256])
    for (int l = 0; l < 2; ++l) {
        int fi = (l == 0) ? FINc : FCc;
        for (int k = 0; k < 4; ++k) w_l[l][k] = fh((size_t)128 * (2 * fi));
        w_l[l][4] = fh(128 * 256);
        w_l[l][5] = fh(128 * 256);
    }
    _Float16* w_gat = fh(4 * 128 * 128);
    _Float16* w_so1 = fh(128 * 128); _Float16* w_so2 = fh(128 * 128);
    _Float16* w_vo1 = fh(128 * 128); _Float16* w_vo2 = fh(128 * 128);
    _Float16* w_fc1 = fh(256 * 512); _Float16* w_fc2 = fh(128 * 256);
    _Float16* w_q = fh(128 * 128); _Float16* w_k = fh(128 * 128); _Float16* w_v = fh(128 * 128);
    _Float16* w_o1 = fh(128 * 384); _Float16* w_o2 = fh(128 * 128);

    // activations
    _Float16* fp16p   = fh((size_t)Bv * KFP);
    float*    z1      = ff(Bv * 512);
    _Float16* z1h     = fh(Bv * 512);
    float*    sc_fpn  = ff(Bv * 128);
    _Float16* atoms16 = fh((size_t)BN * FINc);
    float*    hbuf    = ff((size_t)NHc * BN * FCc);
    _Float16* h16t    = fh((size_t)NHc * BN * FCc);   // per (h,b): [FC,Nn] transposed
    float*    es      = ff(NHc * BN);
    float*    ed      = ff(NHc * BN);
    _Float16* att16   = fh((size_t)NHc * Bv * Nn * Nn);
    float*    ghx     = ff((size_t)NHc * BN * FCc);
    float*    gato    = ff((size_t)BN * FCc);
    float*    tvec    = ff(Bv * 128);
    _Float16* tv16    = fh(Bv * 128);
    float*    t2      = ff(Bv * 128);
    _Float16* t216    = fh(Bv * 128);
    float*    sc_tdim = ff(Bv * 128);
    _Float16* sc16    = fh((size_t)BN * FCc);
    _Float16* vc16    = fh((size_t)BN3 * FCc);
    float* a_ss = ff((size_t)BN * FCc);  float* c_ss = ff((size_t)BN * FCc);
    float* a_sv = ff((size_t)BN * FCc);  float* c_sv = ff((size_t)BN * FCc);
    float* a_vs = ff((size_t)BN3 * FCc); float* c_vs = ff((size_t)BN3 * FCc);
    float* a_vv = ff((size_t)BN3 * FCc); float* c_vv = ff((size_t)BN3 * FCc);
    float*    s_sum   = ff((size_t)BN * 256);
    _Float16* s_sum16 = fh((size_t)BN * 256);
    float*    v_sum   = ff((size_t)BN3 * 256);
    _Float16* v_sum16 = fh((size_t)BN3 * 256);
    float*    sc_new  = ff((size_t)BN * FCc);
    float*    vc_new  = ff((size_t)BN3 * FCc);
    float*    sc_g    = ff(Bv * 128);
    float*    vc_g    = ff(Bv * 384);
    _Float16* g16a    = fh(Bv * 128);
    float*    so_t    = ff(Bv * 128);
    _Float16* so_t16  = fh(Bv * 128);
    float*    sc_out  = ff(Bv * 128);
    _Float16* g16b    = fh(96 * 128);
    float*    vo_t    = ff(96 * 128);
    _Float16* vo_t16  = fh(96 * 128);
    float*    vc_out  = ff(96 * 128);
    _Float16* cat16   = fh(Bv * 512);
    float*    gbuf    = ff(Bv * 256);
    _Float16* g16c    = fh(Bv * 256);
    float*    gcn_out = ff(Bv * 128);
    _Float16* x16     = fh(Bv * 3 * 128);
    float* Qb = ff(96 * 128); float* Kb = ff(96 * 128); float* Vb = ff(96 * 128);
    float*    oatt    = ff(Bv * 384);
    _Float16* oatt16  = fh(Bv * 384);
    float*    ob1     = ff(Bv * 128);
    _Float16* ob1h    = fh(Bv * 128);
    float*    ob2     = ff(Bv * 128);

    auto castk = [&](_Float16* d, const float* s, long n) {
        k_cast<<<dim3((unsigned)((n + 255) / 256)), dim3(256), 0, stream>>>(d, s, n);
    };
    // transpose-cast: src f32 [K,N] (batched) -> dst f16 [N,Kp]
    auto wtrans = [&](_Float16* d, const float* s, int K, int N, int Kp,
                      int batch, long sS, long sD) {
        long tot = (long)N * Kp;
        dim3 g((unsigned)((tot + 255) / 256), 1, batch);
        k_wtrans<<<g, dim3(256), 0, stream>>>(d, s, K, N, Kp, sS, sD);
    };
    auto gemm = [&](const _Float16* A, const _Float16* Bt, const float* bias, float* C,
                    int M, int N, int K, int lda, int ldb, int ldc,
                    long sA, long sB, long sC, int batch, int act) {
        dim3 grid(N / 64, M / 16, batch);
        wmma_gemm<<<grid, dim3(32, 4), 0, stream>>>(A, Bt, bias, C, M, N, K,
                                                    lda, ldb, ldc, sA, sB, sC, act);
    };

    // ---- weight transpose-casts (f32 [K,N] -> f16 Bt [N,Kp]) ----
    wtrans(w_fpn1, P(0), 1489, 512, KFP, 1, 0, 0);
    wtrans(w_fpn2, P(2), 512, 128, 512, 1, 0, 0);
    wtrans(w_tdim, P(4), 64, 128, 64, 4, 64L * 128, 128L * 64);
    wtrans(w_fcg1, P(7), 128, 128, 128, 1, 0, 0);
    wtrans(w_fcg2, P(9), 128, 128, 128, 1, 0, 0);
    for (int l = 0; l < 2; ++l) {
        int pb = 11 + l * 12;
        int fi = (l == 0) ? FINc : FCc;
        wtrans(w_l[l][0], P(pb + 0), 2 * fi, 128, 2 * fi, 1, 0, 0);   // ss_w
        wtrans(w_l[l][1], P(pb + 2), 2 * fi, 128, 2 * fi, 1, 0, 0);   // vs_w
        wtrans(w_l[l][2], P(pb + 4), 2 * fi, 128, 2 * fi, 1, 0, 0);   // sv_w
        wtrans(w_l[l][3], P(pb + 6), 2 * fi, 128, 2 * fi, 1, 0, 0);   // vv_w
        wtrans(w_l[l][4], P(pb + 8), 256, 128, 256, 1, 0, 0);         // cs_w
        wtrans(w_l[l][5], P(pb + 10), 256, 128, 256, 1, 0, 0);        // cv_w
    }
    wtrans(w_gat, P(35), 128, 128, 128, 4, 128L * 128, 128L * 128);
    wtrans(w_so1, P(38), 128, 128, 128, 1, 0, 0);
    wtrans(w_so2, P(40), 128, 128, 128, 1, 0, 0);
    wtrans(w_vo1, P(42), 128, 128, 128, 1, 0, 0);
    wtrans(w_vo2, P(44), 128, 128, 128, 1, 0, 0);
    wtrans(w_fc1, P(46), 512, 256, 512, 1, 0, 0);
    wtrans(w_fc2, P(48), 256, 128, 256, 1, 0, 0);
    wtrans(w_q, P(50), 128, 128, 128, 1, 0, 0);
    wtrans(w_k, P(52), 128, 128, 128, 1, 0, 0);
    wtrans(w_v, P(54), 128, 128, 128, 1, 0, 0);
    wtrans(w_o1, P(56), 384, 128, 384, 1, 0, 0);
    wtrans(w_o2, P(58), 128, 128, 128, 1, 0, 0);

    // ---- input casts ----
    {
        long nt = (long)Bv * KFP;
        k_cast_pad2d<<<dim3((unsigned)((nt + 255) / 256)), 256, 0, stream>>>(fp16p, fp, Bv, 1489, KFP);
    }
    castk(atoms16, atoms, (long)BN * FINc);
    {
        long nz = (long)BN3 * FINc;  // layer-0 vc = zeros (only first fi=64 cols read)
        k_fill_h<<<dim3((unsigned)((nz + 255) / 256)), 256, 0, stream>>>(vc16, nz);
    }

    // ---- FPN branch ----
    gemm(fp16p, w_fpn1, P(1), z1, Bv, 512, KFP, KFP, KFP, 512, 0, 0, 0, 1, ACT_RELU);
    castk(z1h, z1, Bv * 512);
    gemm(z1h, w_fpn2, P(3), sc_fpn, Bv, 128, 512, 512, 512, 128, 0, 0, 0, 1, ACT_NONE);

    // ---- multi-head GAT helper ----
    auto run_gat = [&](const _Float16* X, int K, const _Float16* WhT, int ldw,
                       const float* as_, const float* ad_, float* outBNF) {
        // h[h] = X @ Wh[h] : batched over heads (A shared)
        gemm(X, WhT, nullptr, hbuf, BN, FCc, K, K, ldw, FCc,
             0, (long)FCc * ldw, (long)BN * FCc, NHc, ACT_NONE);
        // per-(h,b) transpose of h [Nn,FC] -> h16t [FC,Nn] (B^T for att@h)
        wtrans(h16t, hbuf, Nn, FCc, Nn, NHc * Bv, (long)Nn * FCc, (long)Nn * FCc);
        k_gat_scores<<<dim3((NHc * BN + 255) / 256), 256, 0, stream>>>(hbuf, as_, ad_, es, ed);
        k_gat_softmax<<<dim3(Nn, Bv, NHc), dim3(64), 0, stream>>>(es, ed, adj, att16);
        // att @ h : batched over (h,b), each 64x64 @ 64x128
        gemm(att16, h16t, nullptr, ghx, Nn, FCc, Nn, Nn, Nn, FCc,
             (long)Nn * Nn, (long)Nn * FCc, (long)Nn * FCc, NHc * Bv, ACT_NONE);
        k_mean_elu<<<dim3((BN * FCc + 255) / 256), 256, 0, stream>>>(ghx, outBNF, BN * FCc);
    };

    // ---- tdim branch ----
    run_gat(atoms16, FINc, w_tdim, 64, P(5), P(6), gato);
    k_sum_n<<<dim3((Bv * FCc + 255) / 256), 256, 0, stream>>>(gato, tvec, Bv, Nn, FCc);
    castk(tv16, tvec, Bv * 128);
    gemm(tv16, w_fcg1, P(8), t2, Bv, 128, 128, 128, 128, 128, 0, 0, 0, 1, ACT_RELU);
    castk(t216, t2, Bv * 128);
    gemm(t216, w_fcg2, P(10), sc_tdim, Bv, 128, 128, 128, 128, 128, 0, 0, 0, 1, ACT_RELU);

    // ---- 3DGCN message-passing layers ----
    const _Float16* scA = atoms16;
    const _Float16* vcA = vc16;
    int fi = FINc;
    for (int l = 0; l < 2; ++l) {
        int pb = 11 + l * 12;
        int ldw = 2 * fi;   // Bt row stride for pair-weight halves
        // pair projections; pair bias folded into the "a" operand.
        // 'a' half: Bt cols 0..fi-1 ; 'c' half: Bt cols fi..2fi-1 (K-offset +fi)
        gemm(scA, w_l[l][0],      P(pb + 1), a_ss, BN,  FCc, fi, fi, ldw, FCc, 0, 0, 0, 1, ACT_NONE);
        gemm(scA, w_l[l][0] + fi, nullptr,   c_ss, BN,  FCc, fi, fi, ldw, FCc, 0, 0, 0, 1, ACT_NONE);
        gemm(vcA, w_l[l][1],      P(pb + 3), a_vs, BN3, FCc, fi, fi, ldw, FCc, 0, 0, 0, 1, ACT_NONE);
        gemm(vcA, w_l[l][1] + fi, nullptr,   c_vs, BN3, FCc, fi, fi, ldw, FCc, 0, 0, 0, 1, ACT_NONE);
        gemm(scA, w_l[l][2],      P(pb + 5), a_sv, BN,  FCc, fi, fi, ldw, FCc, 0, 0, 0, 1, ACT_NONE);
        gemm(scA, w_l[l][2] + fi, nullptr,   c_sv, BN,  FCc, fi, fi, ldw, FCc, 0, 0, 0, 1, ACT_NONE);
        gemm(vcA, w_l[l][3],      P(pb + 7), a_vv, BN3, FCc, fi, fi, ldw, FCc, 0, 0, 0, 1, ACT_NONE);
        gemm(vcA, w_l[l][3] + fi, nullptr,   c_vv, BN3, FCc, fi, fi, ldw, FCc, 0, 0, 0, 1, ACT_NONE);
        // pairwise construct + masked sum over neighbors
        k_pair<<<dim3(BN), dim3(FCc), 0, stream>>>(a_ss, c_ss, a_vs, c_vs, a_sv, c_sv,
                                                   a_vv, c_vv, adj, dists, s_sum, v_sum);
        castk(s_sum16, s_sum, (long)BN * 256);
        castk(v_sum16, v_sum, (long)BN3 * 256);
        gemm(s_sum16, w_l[l][4], P(pb + 9),  sc_new, BN,  FCc, 256, 256, 256, FCc, 0, 0, 0, 1, ACT_RELU);
        gemm(v_sum16, w_l[l][5], P(pb + 11), vc_new, BN3, FCc, 256, 256, 256, FCc, 0, 0, 0, 1, ACT_TANH);
        castk(sc16, sc_new, (long)BN * FCc);
        castk(vc16, vc_new, (long)BN3 * FCc);
        scA = sc16; vcA = vc16; fi = FCc;
    }

    // ---- final GAT + gather ----
    run_gat(sc16, FCc, w_gat, 128, P(36), P(37), gato);
    k_sum_n<<<dim3((Bv * FCc + 255) / 256), 256, 0, stream>>>(gato, sc_g, Bv, Nn, FCc);
    k_sum_n<<<dim3((Bv * 384 + 255) / 256), 256, 0, stream>>>(vc_new, vc_g, Bv, Nn, 384);

    // ---- output MLPs ----
    castk(g16a, sc_g, Bv * 128);
    gemm(g16a, w_so1, P(39), so_t, Bv, 128, 128, 128, 128, 128, 0, 0, 0, 1, ACT_RELU);
    castk(so_t16, so_t, Bv * 128);
    gemm(so_t16, w_so2, P(41), sc_out, Bv, 128, 128, 128, 128, 128, 0, 0, 0, 1, ACT_RELU);

    castk(g16b, vc_g, 96 * 128);
    gemm(g16b, w_vo1, P(43), vo_t, 96, 128, 128, 128, 128, 128, 0, 0, 0, 1, ACT_RELU);
    castk(vo_t16, vo_t, 96 * 128);
    gemm(vo_t16, w_vo2, P(45), vc_out, 96, 128, 128, 128, 128, 128, 0, 0, 0, 1, ACT_RELU);

    k_concat2_cast<<<dim3((Bv * 512 + 255) / 256), 256, 0, stream>>>(cat16, sc_out, vc_out, Bv, 128, 384);
    gemm(cat16, w_fc1, P(47), gbuf, Bv, 256, 512, 512, 512, 256, 0, 0, 0, 1, ACT_RELU);
    castk(g16c, gbuf, Bv * 256);
    gemm(g16c, w_fc2, P(49), gcn_out, Bv, 128, 256, 256, 256, 128, 0, 0, 0, 1, ACT_NONE);

    // ---- 3-token self-attention fusion head ----
    k_stack3_cast<<<dim3((Bv * 3 * 128 + 255) / 256), 256, 0, stream>>>(x16, sc_fpn, sc_tdim, gcn_out, Bv, 128);
    gemm(x16, w_q, P(51), Qb, 96, 128, 128, 128, 128, 128, 0, 0, 0, 1, ACT_NONE);
    gemm(x16, w_k, P(53), Kb, 96, 128, 128, 128, 128, 128, 0, 0, 0, 1, ACT_NONE);
    gemm(x16, w_v, P(55), Vb, 96, 128, 128, 128, 128, 128, 0, 0, 0, 1, ACT_NONE);
    k_selfattn<<<dim3(Bv), dim3(128), 0, stream>>>(Qb, Kb, Vb, oatt);
    castk(oatt16, oatt, Bv * 384);
    gemm(oatt16, w_o1, P(57), ob1, Bv, 128, 384, 384, 384, 128, 0, 0, 0, 1, ACT_RELU);
    castk(ob1h, ob1, Bv * 128);
    gemm(ob1h, w_o2, P(59), ob2, Bv, 128, 128, 128, 128, 128, 0, 0, 0, 1, ACT_RELU);
    k_out<<<dim3(Bv), dim3(128), 0, stream>>>(ob2, P(60), P(61), (float*)d_out);
}
